// VQSign_58514634440711
// MI455X (gfx1250) — compile-verified, matrix operand order checked by reading, not run on hardware
//
#include <hip/hip_runtime.h>
#include <hip/hip_bf16.h>

// ---------------------------------------------------------------------------
// Problem constants (reference: B=32, T=1024, D=1024, S=256, K_FUT=3)
// ---------------------------------------------------------------------------
#define BB 32
#define TT 1024
#define DD 1024
#define SS 256
#define KF 3

typedef __attribute__((ext_vector_type(16))) __bf16 v16bf;
typedef __attribute__((ext_vector_type(8)))  __bf16 v8bf;
typedef __attribute__((ext_vector_type(8)))  float  v8f;

// ---------------------------------------------------------------------------
// WMMA fragment helpers (CDNA5 16x16x32 bf16 layouts, wave32)
// A (16xK, row-major, stride ld):  lane[3:0]=M, lane[4] selects K-half-of-16,
//   v16bf elems 0..7 = K(kh..kh+7), elems 8..15 = K(kh+16..kh+23)
// B (Kx16 supplied as 16 rows x K, i.e. column-major B): identical loader,
//   lane[3:0]=N.
// C/D (16x16 f32): vgpr i -> M = i + 8*lane[4], N = lane[3:0]
// ---------------------------------------------------------------------------
__device__ __forceinline__ v16bf load_frag(const __bf16* __restrict__ p0,
                                           size_t ld, int lane) {
  const __bf16* p = p0 + (size_t)(lane & 15) * ld + (size_t)((lane >> 4) << 3);
  v8bf lo = *reinterpret_cast<const v8bf*>(p);
  v8bf hi = *reinterpret_cast<const v8bf*>(p + 16);
  return __builtin_shufflevector(lo, hi, 0, 1, 2, 3, 4, 5, 6, 7,
                                 8, 9, 10, 11, 12, 13, 14, 15);
}

#define WMMA_BF16(a, b, c)                                              \
  __builtin_amdgcn_wmma_f32_16x16x32_bf16(false, (a), false, (b),       \
                                          (short)0, (c), false, false)

__device__ __forceinline__ float sigmf(float x) {
  return 1.0f / (1.0f + __expf(-x));
}

// ---------------------------------------------------------------------------
// Workspace layout (byte offsets). Total ~418.1 MB.
// ---------------------------------------------------------------------------
#define OFF_WIH  ((size_t)0)                           // 3D*D bf16
#define OFF_WHH  (OFF_WIH + (size_t)3 * DD * DD * 2)   // 3D*D bf16
#define OFF_WP   (OFF_WHH + (size_t)3 * DD * DD * 2)   // D*D bf16
#define OFF_CB   (OFF_WP  + (size_t)DD * DD * 2)       // S*D bf16
#define OFF_CN   (OFF_CB  + (size_t)SS * DD * 2)       // S f32
#define OFF_X    (OFF_CN  + (size_t)1024)              // B*T*D bf16 quantized
#define OFF_H    (OFF_X   + (size_t)BB * TT * DD * 2)  // B*T*D bf16 GRU out
#define OFF_HP   (OFF_H   + (size_t)BB * TT * DD * 2)  // B*T*D bf16 projected
#define OFF_GI   (OFF_HP  + (size_t)BB * TT * DD * 2)  // B*T*3D bf16 gi (all t)
#define OFF_HS   (OFF_GI  + (size_t)BB * TT * 3 * DD * 2)  // B*D f32 h state
#define OFF_HB0  (OFF_HS  + (size_t)BB * DD * 4)       // B*D bf16 h ping
#define OFF_HB1  (OFF_HB0 + (size_t)BB * DD * 2)       // B*D bf16 h pong
#define OFF_SC   (OFF_HB1 + (size_t)BB * DD * 2)       // [0]=cp_sum [1]=mse_sum

// ---------------------------------------------------------------------------
// 0) init: zero h state (f32 + both bf16 ping-pong bufs) and accumulators
// ---------------------------------------------------------------------------
__global__ void k_init(float* h, __bf16* hb0, __bf16* hb1, float* scal) {
  int i = blockIdx.x * blockDim.x + threadIdx.x;
  if (i < BB * DD) {
    h[i] = 0.0f;
    hb0[i] = (__bf16)0.0f;
    hb1[i] = (__bf16)0.0f;
  }
  if (i < 2) scal[i] = 0.0f;
}

// ---------------------------------------------------------------------------
// 1) f32 -> bf16 conversion (weights, codebook)
// ---------------------------------------------------------------------------
__global__ void k_cvt(const float* __restrict__ s, __bf16* __restrict__ d, int n) {
  for (int i = blockIdx.x * blockDim.x + threadIdx.x; i < n;
       i += gridDim.x * blockDim.x)
    d[i] = (__bf16)s[i];
}

// ---------------------------------------------------------------------------
// 2) codebook squared norms (S=256, one thread per code)
// ---------------------------------------------------------------------------
__global__ void k_cnorm(const float* __restrict__ cb, float* __restrict__ cn) {
  int s = threadIdx.x;  // blockDim = 256
  float a = 0.0f;
  const float* r = cb + (size_t)s * DD;
  for (int c = 0; c < DD; ++c) { float v = r[c]; a += v * v; }
  cn[s] = a;
}

// ---------------------------------------------------------------------------
// 3) VQ quantize: per 16-row tile, distance GEMM via WMMA (each wave does two
//    N-tiles sharing the A fragment), packed-key argmin, gather + MSE.
// ---------------------------------------------------------------------------
__global__ __launch_bounds__(256) void k_quant(
    const float* __restrict__ feat, const float* __restrict__ cb,
    const __bf16* __restrict__ cb_bf, const float* __restrict__ cn,
    float* __restrict__ q_out, float* __restrict__ idx_out,
    __bf16* __restrict__ xbf, float* __restrict__ mse_acc) {
  __shared__ __bf16 sA[16 * DD];          // 32 KB feature tile (bf16)
  __shared__ float s_norm[16];
  __shared__ unsigned long long s_best[16];
  __shared__ float s_mse;

  const int tid = threadIdx.x;
  const size_t row0 = (size_t)blockIdx.x * 16;  // flat row = b*T + t

  if (tid < 16) { s_norm[tid] = 0.0f; s_best[tid] = ~0ull; }
  if (tid == 0) s_mse = 0.0f;
  __syncthreads();

  {  // stage tile to LDS as bf16 + row norms
    int r = tid >> 4, c0 = (tid & 15) * 64;
    const float* src = feat + (row0 + r) * DD + c0;
    float acc = 0.0f;
    for (int c = 0; c < 64; ++c) {
      float v = src[c];
      sA[r * DD + c0 + c] = (__bf16)v;
      acc += v * v;
    }
    atomicAdd(&s_norm[r], acc);
  }
  __syncthreads();

  const int wave = tid >> 5, lane = tid & 31;
  {  // 8 waves x 2 N-tiles = 256 codes; A fragment reused across both WMMAs
    const __bf16* b0base = cb_bf + (size_t)wave * 16 * DD;
    const __bf16* b1base = cb_bf + (size_t)(wave + 8) * 16 * DD;
    v8f acc0 = {}, acc1 = {};
    for (int k0 = 0; k0 < DD; k0 += 32) {
      v16bf a = load_frag(&sA[k0], DD, lane);          // ds_load_b128 x2
      v16bf b0 = load_frag(b0base + k0, DD, lane);     // global_load_b128 x2
      v16bf b1 = load_frag(b1base + k0, DD, lane);
      acc0 = WMMA_BF16(a, b0, acc0);
      acc1 = WMMA_BF16(a, b1, acc1);
    }
    const int moff = (lane >> 4) << 3;
    const int n0 = wave * 16 + (lane & 15);
    const int n1 = n0 + 128;
    const float cn0 = cn[n0], cn1 = cn[n1];
    for (int i = 0; i < 8; ++i) {
      float d0 = s_norm[moff + i] - 2.0f * acc0[i] + cn0;
      float d1 = s_norm[moff + i] - 2.0f * acc1[i] + cn1;
      unsigned long long k0v =
          ((unsigned long long)__float_as_uint(d0) << 32) | (unsigned)n0;
      unsigned long long k1v =
          ((unsigned long long)__float_as_uint(d1) << 32) | (unsigned)n1;
      atomicMin(&s_best[moff + i], k0v < k1v ? k0v : k1v);
    }
  }
  __syncthreads();

  {  // gather winner codes, emit f32 + bf16 quantized, MSE partial
    int r = tid >> 4, c0 = (tid & 15) * 64;
    int idx = (int)(s_best[r] & 0xffffffffu);
    size_t orow = (row0 + r) * DD;
    const float* crow = cb + (size_t)idx * DD;
    const float* frow = feat + orow;
    float acc = 0.0f;
    for (int c = 0; c < 64; ++c) {
      float qv = crow[c0 + c];
      q_out[orow + c0 + c] = qv;
      xbf[orow + c0 + c] = (__bf16)qv;
      float dv = frow[c0 + c] - qv;
      acc += dv * dv;
    }
    atomicAdd(&s_mse, acc);
    if ((tid & 15) == 0) idx_out[row0 + r] = (float)idx;
  }
  __syncthreads();
  if (tid == 0) atomicAdd(mse_acc, s_mse);
}

// ---------------------------------------------------------------------------
// 4) gi for ALL timesteps (no sequential dep): gi = X @ Wih^T + b_ih, bf16 out.
//    32768x3072, each wave: 1 M-tile x 2 N-tiles, A fragment reused.
// ---------------------------------------------------------------------------
__global__ __launch_bounds__(256) void k_giall(
    const __bf16* __restrict__ X, const __bf16* __restrict__ Wih,
    const float* __restrict__ bih, __bf16* __restrict__ giall) {
  const int wid = blockIdx.x * 8 + (threadIdx.x >> 5);
  const int lane = threadIdx.x & 31;
  const int mt = wid / 96, ntp = wid % 96;  // 96 N-tile pairs cover 3072 cols
  const __bf16* A = X + (size_t)(mt * 16) * DD;
  const __bf16* B0 = Wih + (size_t)(ntp * 32) * DD;
  const __bf16* B1 = B0 + (size_t)16 * DD;
  v8f acc0 = {}, acc1 = {};
  for (int k0 = 0; k0 < DD; k0 += 32) {
    if (k0 + 256 < DD) __builtin_prefetch(B0 + (size_t)k0 + 256, 0, 1);
    v16bf a = load_frag(A + k0, DD, lane);
    v16bf b0 = load_frag(B0 + k0, DD, lane);
    v16bf b1 = load_frag(B1 + k0, DD, lane);
    acc0 = WMMA_BF16(a, b0, acc0);
    acc1 = WMMA_BF16(a, b1, acc1);
  }
  const int n0 = ntp * 32 + (lane & 15);
  const int moff = (lane >> 4) << 3;
  const float bv0 = bih[n0], bv1 = bih[n0 + 16];
  for (int i = 0; i < 8; ++i) {
    size_t row = (size_t)(mt * 16 + moff + i);
    giall[row * (3 * DD) + n0] = (__bf16)(acc0[i] + bv0);
    giall[row * (3 * DD) + n0 + 16] = (__bf16)(acc1[i] + bv1);
  }
}

// ---------------------------------------------------------------------------
// 5) Fused GRU step: gh = h @ Whh^T (WMMA, K split over 4 waves/tile, LDS
//    reduce) + gate nonlinearity, ONE launch per timestep. h double-buffered
//    (bf16 ping-pong) to avoid cross-block RAW/WAR on the K dimension.
//    Block: jt = 16-column tile of D. 6 tiles (2 M x 3 gates) x 4 K-quarters
//    = 24 waves = 768 threads.
// ---------------------------------------------------------------------------
__global__ __launch_bounds__(768) void k_stepfuse(
    const __bf16* __restrict__ giall, const __bf16* __restrict__ hbin,
    const __bf16* __restrict__ Whh, const float* __restrict__ bhh,
    float* __restrict__ h, __bf16* __restrict__ hbout,
    __bf16* __restrict__ Hbf, int t) {
  __shared__ float s_part[24][16][16];  // 24.5 KB partial gh tiles
  const int jt = blockIdx.x;            // 0..63
  const int wv = threadIdx.x >> 5;      // 0..23
  const int lane = threadIdx.x & 31;
  const int tile = wv >> 2;             // 0..5 : gp*2 + mt
  const int kq = wv & 3;                // K quarter (256 wide)
  const int mt = tile & 1, gp = tile >> 1;

  const __bf16* A = hbin + (size_t)(mt * 16) * DD + (size_t)kq * 256;
  const __bf16* Bb =
      Whh + ((size_t)(gp * DD + jt * 16)) * DD + (size_t)kq * 256;
  v8f acc = {};
  for (int k0 = 0; k0 < 256; k0 += 32) {
    v16bf a = load_frag(A + k0, DD, lane);
    v16bf b = load_frag(Bb + k0, DD, lane);
    acc = WMMA_BF16(a, b, acc);
  }
  {
    const int n = lane & 15, moff = (lane >> 4) << 3;
    for (int i = 0; i < 8; ++i) s_part[wv][moff + i][n] = acc[i];
  }
  __syncthreads();

  const int idx = threadIdx.x;  // 512 gate elements (32 b x 16 cols)
  if (idx < 512) {
    const int b = idx >> 4, jj = idx & 15;
    const int col = jt * 16 + jj;
    const int mtile = b >> 4, m = b & 15;
    float hr = 0.0f, hz = 0.0f, hn_ = 0.0f;
    for (int q = 0; q < 4; ++q) {
      hr += s_part[(0 + mtile) * 4 + q][m][jj];
      hz += s_part[(2 + mtile) * 4 + q][m][jj];
      hn_ += s_part[(4 + mtile) * 4 + q][m][jj];
    }
    hr += bhh[col];
    hz += bhh[DD + col];
    hn_ += bhh[2 * DD + col];
    const size_t girow = ((size_t)b * TT + t) * (size_t)(3 * DD) + col;
    const float r = sigmf((float)giall[girow] + hr);
    const float z = sigmf((float)giall[girow + DD] + hz);
    const float nn = tanhf((float)giall[girow + 2 * DD] + r * hn_);
    const size_t hi = (size_t)b * DD + col;
    const float hnew = (1.0f - z) * nn + z * h[hi];
    h[hi] = hnew;
    hbout[hi] = (__bf16)hnew;
    Hbf[((size_t)b * TT + t) * DD + col] = (__bf16)hnew;
  }
}

// ---------------------------------------------------------------------------
// 6) Projection: Hp = H @ Wp^T + b_p. Each wave: 1 M-tile x 2 N-tiles.
// ---------------------------------------------------------------------------
__global__ __launch_bounds__(256) void k_proj(
    const __bf16* __restrict__ Hbf, const __bf16* __restrict__ Wp,
    const float* __restrict__ bp, __bf16* __restrict__ Hp) {
  const int wid = blockIdx.x * 8 + (threadIdx.x >> 5);
  const int lane = threadIdx.x & 31;
  const int mt = wid >> 5, ntp = wid & 31;  // 32 N-tile pairs cover 1024 cols
  const __bf16* A = Hbf + (size_t)(mt * 16) * DD;
  const __bf16* B0 = Wp + (size_t)(ntp * 32) * DD;
  const __bf16* B1 = B0 + (size_t)16 * DD;
  v8f acc0 = {}, acc1 = {};
  for (int k0 = 0; k0 < DD; k0 += 32) {
    if (k0 + 256 < DD) __builtin_prefetch(B0 + (size_t)k0 + 256, 0, 1);
    v16bf a = load_frag(A + k0, DD, lane);
    v16bf b0 = load_frag(B0 + k0, DD, lane);
    v16bf b1 = load_frag(B1 + k0, DD, lane);
    acc0 = WMMA_BF16(a, b0, acc0);
    acc1 = WMMA_BF16(a, b1, acc1);
  }
  const int n0 = ntp * 32 + (lane & 15);
  const int moff = (lane >> 4) << 3;
  const float bv0 = bp[n0], bv1 = bp[n0 + 16];
  for (int i = 0; i < 8; ++i) {
    size_t row = (size_t)(mt * 16 + moff + i);
    Hp[row * DD + n0] = (__bf16)(acc0[i] + bv0);
    Hp[row * DD + n0 + 16] = (__bf16)(acc1[i] + bv1);
  }
}

// ---------------------------------------------------------------------------
// 7) CPC loss: per (k,t) block: s = sum_b sigmoid(pos_b - neg_b),
//    accumulate -log(s/B) into cp_sum.
// ---------------------------------------------------------------------------
__global__ __launch_bounds__(256) void k_loss(
    const __bf16* __restrict__ Hp, const float* __restrict__ feat,
    const int* __restrict__ negidx, float* __restrict__ cp_acc) {
  const int kb = blockIdx.x / (TT - 1);  // 0..2
  const int t = blockIdx.x % (TT - 1);
  const int k = kb + 1;
  if (t >= TT - k) return;  // uniform per block
  __shared__ float s_sig;
  if (threadIdx.x == 0) s_sig = 0.0f;
  __syncthreads();
  const int wave = threadIdx.x >> 5, lane = threadIdx.x & 31;
  for (int bi = 0; bi < 4; ++bi) {
    const int b = wave * 4 + bi;
    const __bf16* hrow = Hp + ((size_t)b * TT + t) * DD;
    const float* prow = feat + ((size_t)b * TT + (t + k)) * DD;
    const int nb = negidx[((size_t)kb * TT + t) * BB + b];
    const float* nrow = feat + ((size_t)nb * TT + (t + k)) * DD;
    float x = 0.0f;
    for (int d = lane; d < DD; d += 32) {
      float hv = (float)hrow[d];
      x += hv * (prow[d] - nrow[d]);  // pos - neg fused
    }
    for (int off = 16; off; off >>= 1) x += __shfl_down(x, off, 32);
    if (lane == 0) atomicAdd(&s_sig, sigmf(x));
  }
  __syncthreads();
  if (threadIdx.x == 0) atomicAdd(cp_acc, -__logf(s_sig * (1.0f / BB)));
}

// ---------------------------------------------------------------------------
// 8) combine:  total = cp/(K*(T-K)) + (1 + 0.25) * mse_mean
// ---------------------------------------------------------------------------
__global__ void k_final(const float* __restrict__ scal, float* __restrict__ out) {
  float cp = scal[0] / (float)(KF * (TT - KF));
  float mse = scal[1] / (float)((size_t)BB * TT * DD);
  *out = cp + 1.25f * mse;
}

// ---------------------------------------------------------------------------
// launch
// ---------------------------------------------------------------------------
extern "C" void kernel_launch(void* const* d_in, const int* in_sizes, int n_in,
                              void* d_out, int out_size, void* d_ws,
                              size_t ws_size, hipStream_t stream) {
  const float* feat = (const float*)d_in[0];  // (B,T,D)
  const float* cb   = (const float*)d_in[1];  // (S,D)
  const float* Wih  = (const float*)d_in[2];  // (3D,D)
  const float* Whh  = (const float*)d_in[3];  // (3D,D)
  const float* bih  = (const float*)d_in[4];  // (3D,)
  const float* bhh  = (const float*)d_in[5];  // (3D,)
  const float* Wp   = (const float*)d_in[6];  // (D,D)
  const float* bp   = (const float*)d_in[7];  // (D,)
  const int*   nix  = (const int*)d_in[8];    // (K,T,B)

  char* ws = (char*)d_ws;
  __bf16* wih_bf = (__bf16*)(ws + OFF_WIH);
  __bf16* whh_bf = (__bf16*)(ws + OFF_WHH);
  __bf16* wp_bf  = (__bf16*)(ws + OFF_WP);
  __bf16* cb_bf  = (__bf16*)(ws + OFF_CB);
  float*  cn     = (float*)(ws + OFF_CN);
  __bf16* xbf    = (__bf16*)(ws + OFF_X);
  __bf16* hbf    = (__bf16*)(ws + OFF_H);
  __bf16* hpbf   = (__bf16*)(ws + OFF_HP);
  __bf16* giall  = (__bf16*)(ws + OFF_GI);
  float*  hstate = (float*)(ws + OFF_HS);
  __bf16* hb0    = (__bf16*)(ws + OFF_HB0);
  __bf16* hb1    = (__bf16*)(ws + OFF_HB1);
  float*  scal   = (float*)(ws + OFF_SC);

  float* q_out   = (float*)d_out;                 // B*T*D
  float* idx_out = q_out + (size_t)BB * TT * DD;  // B*T
  float* tot_out = idx_out + (size_t)BB * TT;     // 1

  // 0) init state + accumulators
  k_init<<<(BB * DD + 255) / 256, 256, 0, stream>>>(hstate, hb0, hb1, scal);

  // 1) bf16 weight / codebook conversions
  k_cvt<<<1024, 256, 0, stream>>>(Wih, wih_bf, 3 * DD * DD);
  k_cvt<<<1024, 256, 0, stream>>>(Whh, whh_bf, 3 * DD * DD);
  k_cvt<<<1024, 256, 0, stream>>>(Wp, wp_bf, DD * DD);
  k_cvt<<<256, 256, 0, stream>>>(cb, cb_bf, SS * DD);

  // 2) codebook norms
  k_cnorm<<<1, 256, 0, stream>>>(cb, cn);

  // 3) quantize (2048 tiles of 16 rows)
  k_quant<<<(BB * TT) / 16, 256, 0, stream>>>(feat, cb, cb_bf, cn, q_out,
                                              idx_out, xbf, scal + 1);

  // 4) input-side GRU gates for all timesteps (parallel WMMA GEMM)
  k_giall<<<((BB * TT / 16) * 96) / 8, 256, 0, stream>>>(xbf, wih_bf, bih,
                                                         giall);

  // 5) recurrent scan: one fused launch per timestep, ping-pong h
  for (int t = 0; t < TT; ++t) {
    const __bf16* hin = (t & 1) ? hb1 : hb0;
    __bf16* hout = (t & 1) ? hb0 : hb1;
    k_stepfuse<<<DD / 16, 768, 0, stream>>>(giall, hin, whh_bf, bhh, hstate,
                                            hout, hbf, t);
  }

  // 6) projection
  k_proj<<<((BB * TT / 16) * 32) / 8, 256, 0, stream>>>(hbf, wp_bf, bp, hpbf);

  // 7) CPC loss
  k_loss<<<KF * (TT - 1), 256, 0, stream>>>(hpbf, feat, nix, scal);

  // 8) combine
  k_final<<<1, 1, 0, stream>>>(scal, tot_out);
}
